// TransformerDecoder_68771016343795
// MI455X (gfx1250) — compile-verified, matrix operand order checked by reading, not run on hardware
//
#include <hip/hip_runtime.h>
#include <hip/hip_bf16.h>
#include <math.h>

// ---------------------------------------------------------------------------
// Transformer decoder forward for gfx1250 (MI455X), wave32 + WMMA f16.
// All matmuls use v_wmma_f32_16x16x32_f16 with f32 accumulation.
// Round 2: vectorized (b128/b64) tile staging + double-buffered GEMM K-loop.
// ---------------------------------------------------------------------------

typedef _Float16 half_t;
typedef __attribute__((ext_vector_type(16))) _Float16 v16h;
typedef __attribute__((ext_vector_type(4)))  _Float16 v4h;
typedef __attribute__((ext_vector_type(8)))  float    v8f;
typedef __attribute__((ext_vector_type(4)))  float    v4f;

#define N_LAYERS 6
#define N_HEADS  16
#define D_EMBED  1024
#define D_HEAD   64
#define VOCAB    32000
#define FFDIM    4096
#define BATCH    2
#define SEQ      1024
#define M_ROWS   (BATCH * SEQ)        // 2048
#define MH_ROWS  (M_ROWS * N_HEADS)   // 32768 rows of 64 for head projections

// Fragment-ready LDS addressing for 16-bit WMMA operands (ISA 7.12.2):
// 16x32 A tile: lane = (m&15) | (k&8)<<1 ; element = (k&7) | (k&16)>>1.
// B (32x16) uses the symmetric mapping with (n, k).
// Key property: for k 4-aligned, k..k+3 land in 4 consecutive elements of one
// lane => stage with packed v4h (ds_store_b64).
__device__ __forceinline__ int frag_idx(int m, int k) {
  int lane = (m & 15) | ((k & 8) << 1);
  int e    = (k & 7) | ((k & 16) >> 1);
  return lane * 16 + e;
}

__device__ __forceinline__ v8f wmma_f16(v16h a, v16h b, v8f c) {
  return __builtin_amdgcn_wmma_f32_16x16x32_f16(false, a, false, b, (short)0, c,
                                                false, false);
}

// Stage one 128x32 A tile and one 32x64 B tile (fp32 -> f16 fragments).
__device__ __forceinline__ void stage_tiles(
    const float* __restrict__ A, const float* __restrict__ B,
    half_t* lA, half_t* lB, int tb_m, int tb_n, int K, int N, int k0, int tid) {
  // A: 4096 elems as 1024 float4 loads (global_load_b128), 4 per thread.
#pragma unroll
  for (int i = tid; i < 1024; i += 256) {
    int m  = i >> 3;
    int k4 = (i & 7) << 2;
    v4f f = *(const v4f*)&A[(size_t)(tb_m + m) * K + k0 + k4];
    v4h h4;
    h4[0] = (half_t)f[0]; h4[1] = (half_t)f[1];
    h4[2] = (half_t)f[2]; h4[3] = (half_t)f[3];
    *(v4h*)&lA[(m >> 4) * 512 + frag_idx(m, k4)] = h4;  // ds_store_b64
  }
  // B: pack 4 K-rows per thread (coalesced across n), one b64 store each.
#pragma unroll
  for (int i = tid; i < 512; i += 256) {
    int kg = i >> 6;          // 0..7  (k = 4*kg)
    int n  = i & 63;
    const float* bp = &B[(size_t)(k0 + (kg << 2)) * N + tb_n + n];
    v4h h4;
    h4[0] = (half_t)bp[0];
    h4[1] = (half_t)bp[(size_t)N];
    h4[2] = (half_t)bp[2 * (size_t)N];
    h4[3] = (half_t)bp[3 * (size_t)N];
    *(v4h*)&lB[(n >> 4) * 512 + frag_idx(n, kg << 2)] = h4;
  }
}

// ---------------------------------------------------------------------------
// Generic GEMM: C[M,N] = act(A[M,K] @ B[K,N] + bias). A,B fp32 in HBM,
// converted to f16 while staging into LDS; f32 accumulate via WMMA.
// Tile: 128x64 per workgroup (8 waves; wave w owns rows 16w..16w+15).
// Double-buffered staging: stage tile i+1 while WMMAs of tile i execute.
// Requires M%128==0, N%64==0, K%32==0, 16B-aligned A (true for all shapes).
// ---------------------------------------------------------------------------
__global__ __launch_bounds__(256) void gemm_wmma_kernel(
    const float* __restrict__ A, const float* __restrict__ B,
    const float* __restrict__ bias, float* __restrict__ C,
    int M, int N, int K, int relu) {
  __shared__ __attribute__((aligned(32))) half_t lA[2][8 * 512]; // 2x 8KB
  __shared__ __attribute__((aligned(32))) half_t lB[2][4 * 512]; // 2x 4KB

  const int tb_m = blockIdx.y * 128;
  const int tb_n = blockIdx.x * 64;
  const int tid  = threadIdx.x;
  const int wave = tid >> 5;
  const int lane = tid & 31;

  v8f acc[4];
#pragma unroll
  for (int i = 0; i < 4; ++i)
#pragma unroll
    for (int j = 0; j < 8; ++j) acc[i][j] = 0.0f;

  stage_tiles(A, B, lA[0], lB[0], tb_m, tb_n, K, N, 0, tid);

  int cur = 0;
  for (int k0 = 0; k0 < K; k0 += 32) {
    __syncthreads();  // staging of `cur` complete; prior reads of `cur^1` done
    if (k0 + 32 < K) {
      stage_tiles(A, B, lA[cur ^ 1], lB[cur ^ 1], tb_m, tb_n, K, N, k0 + 32, tid);
      // hint the strip after next toward L2 (global_prefetch_b8)
      if (k0 + 64 < K)
        __builtin_prefetch(&A[(size_t)(tb_m + (tid >> 1)) * K + k0 + 64 + ((tid & 1) << 4)], 0, 3);
    }
    v16h afrag = *(const v16h*)&lA[cur][wave * 512 + lane * 16];
#pragma unroll
    for (int nb = 0; nb < 4; ++nb) {
      v16h bfrag = *(const v16h*)&lB[cur][nb * 512 + lane * 16];
      acc[nb] = wmma_f16(afrag, bfrag, acc[nb]);
    }
    cur ^= 1;
  }

  // C layout (ISA 7.12.2): VGPR j -> row j (lanes 0-15) / row j+8 (lanes 16-31).
  const int rbase = tb_m + wave * 16 + ((lane >> 4) << 3);
  const int col0  = lane & 15;
#pragma unroll
  for (int nb = 0; nb < 4; ++nb) {
    int col  = tb_n + nb * 16 + col0;
    float bv = bias ? bias[col] : 0.0f;
#pragma unroll
    for (int j = 0; j < 8; ++j) {
      float vv = acc[nb][j] + bv;
      if (relu) vv = fmaxf(vv, 0.0f);
      C[(size_t)(rbase + j) * N + col] = vv;
    }
  }
}

// ---------------------------------------------------------------------------
// Attention: one wave per (b, h, 16-query tile). q/k/v are [B,S,H,64] fp32.
// Scores via WMMA (K-dim 64 => 2 chained 16x16x32), softmax over full row
// (1024 keys, includes the reference's 1e-19 masked-score quirk), then P@V.
// ---------------------------------------------------------------------------
__global__ __launch_bounds__(32) void attn_wmma_kernel(
    const float* __restrict__ q, const float* __restrict__ k,
    const float* __restrict__ v, float* __restrict__ out, int causal) {
  __shared__ __attribute__((aligned(32))) half_t sc[16 * SEQ];  // scores (half)
  __shared__ __attribute__((aligned(32))) half_t lQ[2 * 512];   // 16x64 Q frags
  __shared__ __attribute__((aligned(32))) half_t lKV[4 * 512];  // K/V staging
  __shared__ __attribute__((aligned(32))) half_t pfrag[512];    // prob frag
  __shared__ float rowmax[16];
  __shared__ float rowinv[16];
  __shared__ float red[32];

  const int b  = blockIdx.z;
  const int h  = blockIdx.y;
  const int q0 = blockIdx.x * 16;
  const int lane = threadIdx.x;

  // Stage Q tile [16 x 64]: b128 loads + packed b64 fragment stores.
#pragma unroll
  for (int i = lane; i < 256; i += 32) {
    int m  = i >> 4;
    int d4 = (i & 15) << 2;
    v4f f = *(const v4f*)&q[(size_t)(((b * SEQ + q0 + m) * N_HEADS + h)) * D_HEAD + d4];
    v4h h4;
    h4[0] = (half_t)f[0]; h4[1] = (half_t)f[1];
    h4[2] = (half_t)f[2]; h4[3] = (half_t)f[3];
    *(v4h*)&lQ[(d4 >> 5) * 512 + frag_idx(m, d4 & 31)] = h4;
  }
  __syncthreads();
  v16h qf0 = *(const v16h*)&lQ[0 * 512 + lane * 16];
  v16h qf1 = *(const v16h*)&lQ[1 * 512 + lane * 16];

  const int rowoff = (lane >> 4) << 3;
  const int col    = lane & 15;

  // ---- Score pass: 64 key tiles of 16 ----
  for (int kt = 0; kt < SEQ / 16; ++kt) {
#pragma unroll
    for (int i = lane; i < 256; i += 32) { // stage K tile as B-fragments
      int m  = i >> 4;
      int d4 = (i & 15) << 2;
      v4f f = *(const v4f*)&k[(size_t)(((b * SEQ + kt * 16 + m) * N_HEADS + h)) * D_HEAD + d4];
      v4h h4;
      h4[0] = (half_t)f[0]; h4[1] = (half_t)f[1];
      h4[2] = (half_t)f[2]; h4[3] = (half_t)f[3];
      *(v4h*)&lKV[(d4 >> 5) * 512 + frag_idx(m, d4 & 31)] = h4;
    }
    __syncthreads();
    v8f s8;
#pragma unroll
    for (int j = 0; j < 8; ++j) s8[j] = 0.0f;
    s8 = wmma_f16(qf0, *(const v16h*)&lKV[0 * 512 + lane * 16], s8);
    s8 = wmma_f16(qf1, *(const v16h*)&lKV[1 * 512 + lane * 16], s8);
    __syncthreads();
    int key = kt * 16 + col;
#pragma unroll
    for (int j = 0; j < 8; ++j) {
      int qrow = rowoff + j;
      float sv = s8[j] * 0.125f; // 1/sqrt(64)
      if (causal && key > q0 + qrow) sv = 1e-19f; // reference quirk: not -inf
      sc[qrow * SEQ + key] = (half_t)sv;
    }
    __syncthreads();
  }

  // ---- Softmax stats per row ----
  for (int r = 0; r < 16; ++r) {
    float mx = -1e30f;
    for (int i = lane; i < SEQ; i += 32) mx = fmaxf(mx, (float)sc[r * SEQ + i]);
    red[lane] = mx;
    __syncthreads();
    if (lane == 0) {
      float m2 = red[0];
      for (int t = 1; t < 32; ++t) m2 = fmaxf(m2, red[t]);
      rowmax[r] = m2;
    }
    __syncthreads();
    float m2 = rowmax[r];
    float sm = 0.0f;
    for (int i = lane; i < SEQ; i += 32) sm += __expf((float)sc[r * SEQ + i] - m2);
    red[lane] = sm;
    __syncthreads();
    if (lane == 0) {
      float t2 = 0.0f;
      for (int t = 0; t < 32; ++t) t2 += red[t];
      rowinv[r] = 1.0f / t2;
    }
    __syncthreads();
  }

  // ---- P @ V ----
  v8f o[4];
#pragma unroll
  for (int i = 0; i < 4; ++i)
#pragma unroll
    for (int j = 0; j < 8; ++j) o[i][j] = 0.0f;

  for (int c = 0; c < SEQ / 32; ++c) {
    // Stage V chunk [32 keys x 64 dims]: pack 4 keys per b64 store.
#pragma unroll
    for (int i = lane; i < 512; i += 32) {
      int kg = i >> 6;        // 0..7  (key = 4*kg within chunk)
      int d  = i & 63;
      const float* vp = &v[(size_t)(((b * SEQ + c * 32 + (kg << 2)) * N_HEADS + h)) * D_HEAD + d];
      const size_t rs = (size_t)N_HEADS * D_HEAD; // row stride
      v4h h4;
      h4[0] = (half_t)vp[0];
      h4[1] = (half_t)vp[rs];
      h4[2] = (half_t)vp[2 * rs];
      h4[3] = (half_t)vp[3 * rs];
      *(v4h*)&lKV[(d >> 4) * 512 + frag_idx(d, kg << 2)] = h4;
    }
    // Prob fragment: 4 keys per packed store.
#pragma unroll
    for (int i = lane; i < 128; i += 32) {
      int m  = i >> 3;
      int k4 = (i & 7) << 2;
      float mx = rowmax[m], iv = rowinv[m];
      v4h h4;
#pragma unroll
      for (int j = 0; j < 4; ++j) {
        float s = (float)sc[m * SEQ + c * 32 + k4 + j];
        h4[j] = (half_t)(__expf(s - mx) * iv);
      }
      *(v4h*)&pfrag[frag_idx(m, k4)] = h4;
    }
    __syncthreads();
    v16h pf = *(const v16h*)&pfrag[lane * 16];
#pragma unroll
    for (int nb = 0; nb < 4; ++nb)
      o[nb] = wmma_f16(pf, *(const v16h*)&lKV[nb * 512 + lane * 16], o[nb]);
    __syncthreads();
  }

#pragma unroll
  for (int nb = 0; nb < 4; ++nb)
#pragma unroll
    for (int j = 0; j < 8; ++j)
      out[(size_t)(((b * SEQ + q0 + rowoff + j) * N_HEADS + h)) * D_HEAD +
          nb * 16 + col] = o[nb][j];
}

// ---------------------------------------------------------------------------
// out = LayerNorm(a + b) * g + beta   (one 256-thread block per row)
// ---------------------------------------------------------------------------
__global__ __launch_bounds__(256) void add_ln_kernel(
    const float* __restrict__ a, const float* __restrict__ bres,
    const float* __restrict__ g, const float* __restrict__ be,
    float* __restrict__ out) {
  __shared__ float rs[256];
  __shared__ float rs2[256];
  const int row = blockIdx.x;
  const int tid = threadIdx.x;
  const float* pa = a + (size_t)row * D_EMBED;
  const float* pb = bres + (size_t)row * D_EMBED;
  float s = 0.0f, s2 = 0.0f;
  for (int i = tid; i < D_EMBED; i += 256) {
    float x = pa[i] + pb[i];
    s += x; s2 += x * x;
  }
  rs[tid] = s; rs2[tid] = s2;
  __syncthreads();
  for (int off = 128; off > 0; off >>= 1) {
    if (tid < off) { rs[tid] += rs[tid + off]; rs2[tid] += rs2[tid + off]; }
    __syncthreads();
  }
  float mu  = rs[0] / (float)D_EMBED;
  float var = rs2[0] / (float)D_EMBED - mu * mu;
  float inv = rsqrtf(var + 1e-5f);
  for (int i = tid; i < D_EMBED; i += 256) {
    float x = pa[i] + pb[i];
    out[(size_t)row * D_EMBED + i] = (x - mu) * inv * g[i] + be[i];
  }
}

// ---------------------------------------------------------------------------
// x = emb[ids] * sqrt(D) + pos_encoding (reference quirk: sin if d even else
// cos of the SAME angle pos / 10000^(2d/D))
// ---------------------------------------------------------------------------
__global__ __launch_bounds__(256) void embed_pos_kernel(
    const int* __restrict__ ids, const float* __restrict__ emb,
    float* __restrict__ x) {
  int idx = blockIdx.x * 256 + threadIdx.x;
  if (idx >= M_ROWS * D_EMBED) return;
  int m = idx / D_EMBED;
  int d = idx - m * D_EMBED;
  int s = m % SEQ;
  float val = emb[(size_t)ids[m] * D_EMBED + d] * 32.0f; // sqrt(1024)
  float angle = (float)s * powf(10000.0f, -2.0f * (float)d / (float)D_EMBED);
  val += ((d & 1) == 0) ? sinf(angle) : cosf(angle);
  x[idx] = val;
}

// ---------------------------------------------------------------------------
// Host-side orchestration (graph-capture safe: launches only).
// ---------------------------------------------------------------------------
extern "C" void kernel_launch(void* const* d_in, const int* in_sizes, int n_in,
                              void* d_out, int out_size, void* d_ws, size_t ws_size,
                              hipStream_t stream) {
  (void)in_sizes; (void)n_in; (void)out_size; (void)ws_size;

  const int*   ids   = (const int*)  d_in[0];
  const float* enc_k = (const float*)d_in[1];
  const float* enc_v = (const float*)d_in[2];
  const float* emb   = (const float*)d_in[3];
  const float* Wq_m  = (const float*)d_in[4];
  const float* Wk_m  = (const float*)d_in[5];
  const float* Wv_m  = (const float*)d_in[6];
  const float* Wq_c  = (const float*)d_in[7];
  const float* Wk_c  = (const float*)d_in[8];
  const float* Wv_c  = (const float*)d_in[9];
  const float* ln1_g = (const float*)d_in[10];
  const float* ln1_b = (const float*)d_in[11];
  const float* ln2_g = (const float*)d_in[12];
  const float* ln2_b = (const float*)d_in[13];
  const float* ln3_g = (const float*)d_in[14];
  const float* ln3_b = (const float*)d_in[15];
  const float* W1    = (const float*)d_in[16];
  const float* b1    = (const float*)d_in[17];
  const float* W2    = (const float*)d_in[18];
  const float* b2    = (const float*)d_in[19];
  const float* Wout  = (const float*)d_in[20];
  const float* bout  = (const float*)d_in[21];
  float* outp = (float*)d_out;

  // Workspace layout (fp32): needs (7*M*D + M*FF)*4 = ~88 MB.
  float* ws   = (float*)d_ws;
  float* x    = ws;                                // [M, D]
  float* qb   = x    + (size_t)M_ROWS * D_EMBED;   // [M, D]
  float* kb   = qb   + (size_t)M_ROWS * D_EMBED;   // [M, D]
  float* vb   = kb   + (size_t)M_ROWS * D_EMBED;   // [M, D]
  float* attn = vb   + (size_t)M_ROWS * D_EMBED;   // [M, D] (also ff2)
  float* h1   = attn + (size_t)M_ROWS * D_EMBED;   // [M, D]
  float* h2   = h1   + (size_t)M_ROWS * D_EMBED;   // [M, D]
  float* ff1  = h2   + (size_t)M_ROWS * D_EMBED;   // [M, FF]

  const dim3 gProj(D_HEAD / 64, MH_ROWS / 128);    // (1, 256)
  const dim3 gAttn(SEQ / 16, N_HEADS, BATCH);      // (64, 16, 2)
  const dim3 gFF1(FFDIM / 64, M_ROWS / 128);       // (64, 16)
  const dim3 gFF2(D_EMBED / 64, M_ROWS / 128);     // (16, 16)
  const dim3 gOut(VOCAB / 64, M_ROWS / 128);       // (500, 16)

  embed_pos_kernel<<<(M_ROWS * D_EMBED + 255) / 256, 256, 0, stream>>>(ids, emb, x);

  for (int l = 0; l < N_LAYERS; ++l) {
    const size_t wo = (size_t)l * D_HEAD * D_HEAD;
    // ---- masked self-attention ----
    gemm_wmma_kernel<<<gProj, 256, 0, stream>>>(x, Wq_m + wo, nullptr, qb, MH_ROWS, D_HEAD, D_HEAD, 0);
    gemm_wmma_kernel<<<gProj, 256, 0, stream>>>(x, Wk_m + wo, nullptr, kb, MH_ROWS, D_HEAD, D_HEAD, 0);
    gemm_wmma_kernel<<<gProj, 256, 0, stream>>>(x, Wv_m + wo, nullptr, vb, MH_ROWS, D_HEAD, D_HEAD, 0);
    attn_wmma_kernel<<<gAttn, 32, 0, stream>>>(qb, kb, vb, attn, 1);
    add_ln_kernel<<<M_ROWS, 256, 0, stream>>>(attn, x, ln1_g + l * D_EMBED, ln1_b + l * D_EMBED, h1);
    // ---- cross-attention (k/v from encoder) ----
    gemm_wmma_kernel<<<gProj, 256, 0, stream>>>(h1,    Wq_c + wo, nullptr, qb, MH_ROWS, D_HEAD, D_HEAD, 0);
    gemm_wmma_kernel<<<gProj, 256, 0, stream>>>(enc_k, Wk_c + wo, nullptr, kb, MH_ROWS, D_HEAD, D_HEAD, 0);
    gemm_wmma_kernel<<<gProj, 256, 0, stream>>>(enc_v, Wv_c + wo, nullptr, vb, MH_ROWS, D_HEAD, D_HEAD, 0);
    attn_wmma_kernel<<<gAttn, 32, 0, stream>>>(qb, kb, vb, attn, 0);
    add_ln_kernel<<<M_ROWS, 256, 0, stream>>>(attn, h1, ln2_g + l * D_EMBED, ln2_b + l * D_EMBED, h2);
    // ---- FFN ----
    gemm_wmma_kernel<<<gFF1, 256, 0, stream>>>(h2,  W1 + (size_t)l * D_EMBED * FFDIM, b1 + (size_t)l * FFDIM,
                                               ff1, M_ROWS, FFDIM, D_EMBED, 1);
    gemm_wmma_kernel<<<gFF2, 256, 0, stream>>>(ff1, W2 + (size_t)l * FFDIM * D_EMBED, b2 + (size_t)l * D_EMBED,
                                               attn, M_ROWS, D_EMBED, FFDIM, 0);
    add_ln_kernel<<<M_ROWS, 256, 0, stream>>>(attn, h2, ln3_g + l * D_EMBED, ln3_b + l * D_EMBED, x);
  }

  // ---- final vocab projection ----
  gemm_wmma_kernel<<<gOut, 256, 0, stream>>>(x, Wout, bout, outp, M_ROWS, VOCAB, D_EMBED, 0);
}
